// SpatialModule_4183298146531
// MI455X (gfx1250) — matrix operand order repaired and checked by reading, not compiled, not run on hardware
//
#include <hip/hip_runtime.h>
#include <hip/hip_bf16.h>
#include <math.h>

// ---------------------------------------------------------------------------
// CDNA5 (gfx1250) implementation.
//  Stage 1: DFT-as-GEMM with V_WMMA_F32_16X16X32_F16 (f16 in, f32 acc),
//           x staged once (coalesced) into LDS as f16; A-fragments via
//           ds_load_b128; spectra spilled as packed f16 (Re,Im) pairs.
//  Stage 2: per-group cross-spectral stats -> A (3x3 antisymmetric / den)
//  Stage 3: tiny attention per (b,s,group)
//  Stage 4: batchnorm stats over (b,c) with LDS float atomics (ds_add_f32)
//  Stage 5: bn-apply + FC + transpose to (b, 21, 384)
// ---------------------------------------------------------------------------

typedef __attribute__((ext_vector_type(16))) _Float16 v16h;
typedef __attribute__((ext_vector_type(8)))  _Float16 v8h;
typedef __attribute__((ext_vector_type(4)))  _Float16 v4h;
typedef __attribute__((ext_vector_type(2)))  _Float16 v2h;
typedef __attribute__((ext_vector_type(8)))  float    v8f;

#define SD            256
#define B_DIM         64
#define S_DIM         128
#define NGRP_TOTAL    (B_DIM * S_DIM * 7)          /* 57344 3-channel groups */
#define GROUPS_PER_WG 16
#define ROWS_PER_WG   48                           /* 16 groups * 3 channels */
#define NWG_DFT       (NGRP_TOTAL / GROUPS_PER_WG) /* 3584 */
#define AP            272                          /* lA row pitch (halfs)   */
#define LPH           136                          /* lRI row pitch (pairs)  */

__constant__ int dCOMB[7][3] = {
    {0, 1, 2}, {1, 0, 4}, {2, 0, 5}, {3, 2, 4}, {4, 1, 6}, {5, 3, 6}, {6, 3, 5}};

// ---------------------------------------------------------------------------
// Kernel 0: build transposed DFT matrices in f16.
//   btc[f*256+k] =  cos(2*pi*f*k/256)   (Re weights)
//   bts[f*256+k] = -sin(2*pi*f*k/256)   (Im weights, e^{-i theta})
// Integer phase (f*k mod 256) keeps the angle exact.
// ---------------------------------------------------------------------------
__global__ __launch_bounds__(256) void k_build_dft(_Float16* __restrict__ btc,
                                                   _Float16* __restrict__ bts) {
  const int tid = blockIdx.x * blockDim.x + threadIdx.x;  // 0..65535
  const int f = tid >> 8, k = tid & 255;
  const float TWO_PI_OVER_256 = 0.0245436926061702596f;
  float ang = TWO_PI_OVER_256 * (float)((f * k) & 255);
  float sv, cv;
  __sincosf(ang, &sv, &cv);
  btc[tid] = (_Float16)cv;
  bts[tid] = (_Float16)(-sv);
}

// ---------------------------------------------------------------------------
// Kernel 1: DFT via WMMA + per-group cross-spectral statistics.
// Block = 256 threads (8 wave32) owning rows [48*blk, 48*blk+48) of the
// (172032 x 256) fp32 "x" matrix (3 consecutive rows = one channel group).
//
//  - x slab (48 KB) is read ONCE, coalesced float4, packed-converted to f16
//    into LDS (lA).  HBM roofline: one pass over x = ~8 us on MI455X.
//  - WMMA A-fragments come from LDS via two ds_load_b128 per fragment;
//    B-fragments (DFT cos/-sin, 256 KB total) stay L2-resident across all
//    3584 workgroups and load as single 32B vector loads.
//  - Each wave owns one 16-freq tile per 128-freq pass; accumulators are
//    spilled to LDS as packed f16 (Re,Im) pairs (single ds_store_b32 each).
// LDS: 48*272*2 (lA) + 48*136*4 (lRI) = 52,224 B.
// ---------------------------------------------------------------------------
__global__ __launch_bounds__(256) void k_dft_stats(const float* __restrict__ x,
                                                   const _Float16* __restrict__ btc,
                                                   const _Float16* __restrict__ bts,
                                                   float* __restrict__ Aout) {
  __shared__ _Float16 lA[ROWS_PER_WG * AP];        // f16 time-domain rows
  __shared__ _Float16 lRI[ROWS_PER_WG * LPH * 2];  // packed (Re,Im) f16 pairs

  const int t    = threadIdx.x;
  const int lane = t & 31;
  const int wave = t >> 5;     // 0..7 -> one 16-freq tile per pass
  const int half = lane >> 4;  // wave32 upper/lower 16 lanes
  const int l16  = lane & 15;
  const int row0 = blockIdx.x * ROWS_PER_WG;

  // ---- stage x slab into LDS: coalesced float4 + v_cvt_pk_f16_f32 ---------
  {
    const float4* xs = (const float4*)(x + (size_t)row0 * SD);
#pragma unroll
    for (int i = 0; i < 12; ++i) {
      const int q = t + i * 256;      // 0..3071 (48 rows * 64 float4)
      float4 f = xs[q];
      const int row = q >> 6, col = (q & 63) * 4;
      v4h h;
      h[0] = (_Float16)f.x; h[1] = (_Float16)f.y;
      h[2] = (_Float16)f.z; h[3] = (_Float16)f.w;
      *(v4h*)(lA + row * AP + col) = h;   // ds_store_b64
    }
  }
  __syncthreads();

  // per-thread partial stats (group = t>>4, 16 freq-lane slots each)
  float n01 = 0.f, n02 = 0.f, n12 = 0.f;
  float d01 = 0.f, d02 = 0.f, d12 = 0.f;

  for (int pass = 0; pass < 2; ++pass) {
    // -------- WMMA phase: wave `w` computes freqs [pass*128+w*16, +16) -----
    {
      const int ntl   = wave;
      const int fbase = pass * 128 + ntl * 16;

      v8f accRe[3] = {};
      v8f accIm[3] = {};

      // B rows: lane -> freq column (N = lane%16)
      const _Float16* bcrow = btc + (size_t)(fbase + l16) * SD;
      const _Float16* bsrow = bts + (size_t)(fbase + l16) * SD;

#pragma unroll 4
      for (int ks = 0; ks < 8; ++ks) {  // K = 256 in 8 steps of 32
        // B layout (32x16 f16): lane holds K = 16*half + e, e = 0..15
        const int kb = ks * 32 + half * 16;
        v16h bfc = *(const v16h*)(bcrow + kb);  // 32B vector load
        v16h bfs = *(const v16h*)(bsrow + kb);

#pragma unroll
        for (int rt = 0; rt < 3; ++rt) {
          // A layout (16x32 f16): lane row M = lane%16,
          //   elems 0..15 -> K = ks*32 + 16*half + e
          const int abase = (rt * 16 + l16) * AP + ks * 32 + half * 16;
          v8h lo = *(const v8h*)(lA + abase);        // ds_load_b128
          v8h hi = *(const v8h*)(lA + abase + 8);    // ds_load_b128
          v16h a;
#pragma unroll
          for (int e = 0; e < 8; ++e) { a[e] = lo[e]; a[8 + e] = hi[e]; }

          accRe[rt] = __builtin_amdgcn_wmma_f32_16x16x32_f16(
              false, a, false, bfc, (short)0, accRe[rt], false, false);
          accIm[rt] = __builtin_amdgcn_wmma_f32_16x16x32_f16(
              false, a, false, bfs, (short)0, accIm[rt], false, false);
        }
      }

      // D layout (16x16 f32): VGPR v -> row M = v + 8*half, col N = lane%16.
      // Pack (Re,Im) to f16x2, single ds_store_b32 per element.
#pragma unroll
      for (int rt = 0; rt < 3; ++rt) {
#pragma unroll
        for (int v = 0; v < 8; ++v) {
          const int r = rt * 16 + v + half * 8;
          const int c = ntl * 16 + l16;
          v2h p;
          p[0] = (_Float16)accRe[rt][v];
          p[1] = (_Float16)accIm[rt][v];
          *(v2h*)(lRI + (r * LPH + c) * 2) = p;
        }
      }
    }
    __syncthreads();

    // -------- stats phase over this 128-freq half --------------------------
    {
      const int g     = t >> 4;  // group 0..15
      const int s16   = t & 15;  // freq-lane slot
      const int rbase = g * 3;
#pragma unroll
      for (int jj = 0; jj < 8; ++jj) {
        const int fl = s16 * 8 + jj;
        v2h p0 = *(const v2h*)(lRI + ((rbase + 0) * LPH + fl) * 2);
        v2h p1 = *(const v2h*)(lRI + ((rbase + 1) * LPH + fl) * 2);
        v2h p2 = *(const v2h*)(lRI + ((rbase + 2) * LPH + fl) * 2);
        float r0 = (float)p0[0], i0 = (float)p0[1];
        float r1 = (float)p1[0], i1 = (float)p1[1];
        float r2 = (float)p2[0], i2 = (float)p2[1];
        float m0 = sqrtf(r0 * r0 + i0 * i0);
        float m1 = sqrtf(r1 * r1 + i1 * i1);
        float m2 = sqrtf(r2 * r2 + i2 * i2);
        n01 += i0 * r1 - r0 * i1;  // Im(F0 * conj(F1))
        n02 += i0 * r2 - r0 * i2;
        n12 += i1 * r2 - r1 * i2;
        d01 += m0 * m1;
        d02 += m0 * m2;
        d12 += m1 * m2;
      }
    }
    __syncthreads();
  }

  // reduce the 16 freq-lane partials of each group (width-16 xor shuffle)
#pragma unroll
  for (int off = 1; off < 16; off <<= 1) {
    n01 += __shfl_xor(n01, off, 16);
    n02 += __shfl_xor(n02, off, 16);
    n12 += __shfl_xor(n12, off, 16);
    d01 += __shfl_xor(d01, off, 16);
    d02 += __shfl_xor(d02, off, 16);
    d12 += __shfl_xor(d12, off, 16);
  }

  if ((t & 15) == 0) {
    const int g = t >> 4;
    // A = num/den with zeroed diagonal; num antisymmetric, den symmetric.
    // (common 1/sd factors cancel in the ratio)
    float a01 = n01 / d01, a02 = n02 / d02, a12 = n12 / d12;
    float* o = Aout + (size_t)(blockIdx.x * GROUPS_PER_WG + g) * 9;
    o[0] = 0.f;   o[1] = a01;  o[2] = a02;
    o[3] = -a01;  o[4] = 0.f;  o[5] = a12;
    o[6] = -a02;  o[7] = -a12; o[8] = 0.f;
  }
}

// ---------------------------------------------------------------------------
// Kernel 2: per-(b,s,outgroup) attention. All 3x3 work in registers.
// ---------------------------------------------------------------------------
__device__ __forceinline__ void lin3(const float* T, const float* W,
                                     const float* b, float* O) {
#pragma unroll
  for (int r = 0; r < 3; ++r)
#pragma unroll
    for (int o = 0; o < 3; ++o)
      O[r * 3 + o] = T[r * 3 + 0] * W[o * 3 + 0] + T[r * 3 + 1] * W[o * 3 + 1] +
                     T[r * 3 + 2] * W[o * 3 + 2] + b[o];
}

__device__ __forceinline__ void attn3(const float* Q, const float* K,
                                      const float* V, float* O) {
  const float is3 = 0.57735026918962576f;  // 1/sqrt(3)
#pragma unroll
  for (int r = 0; r < 3; ++r) {
    float s0 = (Q[r*3] * K[0] + Q[r*3+1] * K[1] + Q[r*3+2] * K[2]) * is3;
    float s1 = (Q[r*3] * K[3] + Q[r*3+1] * K[4] + Q[r*3+2] * K[5]) * is3;
    float s2 = (Q[r*3] * K[6] + Q[r*3+1] * K[7] + Q[r*3+2] * K[8]) * is3;
    float m  = fmaxf(s0, fmaxf(s1, s2));
    float w0 = __expf(s0 - m), w1 = __expf(s1 - m), w2 = __expf(s2 - m);
    float inv = 1.f / (w0 + w1 + w2);
    w0 *= inv; w1 *= inv; w2 *= inv;
    O[r*3+0] = w0 * V[0] + w1 * V[3] + w2 * V[6];
    O[r*3+1] = w0 * V[1] + w1 * V[4] + w2 * V[7];
    O[r*3+2] = w0 * V[2] + w1 * V[5] + w2 * V[8];
  }
}

__global__ __launch_bounds__(256) void k_attn(
    const float* __restrict__ A, const float* __restrict__ Wq,
    const float* __restrict__ bq, const float* __restrict__ Wk,
    const float* __restrict__ bk, const float* __restrict__ Wv,
    const float* __restrict__ bv, float* __restrict__ att) {
  const int idx = blockIdx.x * blockDim.x + threadIdx.x;
  if (idx >= NGRP_TOTAL) return;
  const int n = idx % 7, bs = idx / 7;

  const float* Ai = A + (size_t)(bs * 7 + dCOMB[n][0]) * 9;
  const float* Aj = A + (size_t)(bs * 7 + dCOMB[n][1]) * 9;
  const float* Ak = A + (size_t)(bs * 7 + dCOMB[n][2]) * 9;

  float Qa[9], Ka[9], Va[9], Qb[9], Kb[9], Vb[9], Qc[9], Kc[9], Vc[9];
  lin3(Ai, Wq + 0,  bq + 0, Qa);  lin3(Ai, Wk + 0,  bk + 0, Ka);  lin3(Ai, Wv + 0,  bv + 0, Va);
  lin3(Aj, Wq + 9,  bq + 3, Qb);  lin3(Aj, Wk + 9,  bk + 3, Kb);  lin3(Aj, Wv + 9,  bv + 3, Vb);
  lin3(Ak, Wq + 18, bq + 6, Qc);  lin3(Ak, Wk + 18, bk + 6, Kc);  lin3(Ak, Wv + 18, bv + 6, Vc);

  float P0[9], P1[9], P2[9];
  attn3(Qc, Ka, Vb, P0);
  attn3(Qa, Kb, Vc, P1);
  attn3(Qb, Kc, Va, P2);

  float* o = att + (size_t)idx * 27;
#pragma unroll
  for (int r = 0; r < 3; ++r) {
    o[r*9+0] = P0[r*3+0]; o[r*9+1] = P0[r*3+1]; o[r*9+2] = P0[r*3+2];
    o[r*9+3] = P1[r*3+0]; o[r*9+4] = P1[r*3+1]; o[r*9+5] = P1[r*3+2];
    o[r*9+6] = P2[r*3+0]; o[r*9+7] = P2[r*3+1]; o[r*9+8] = P2[r*3+2];
  }
}

// ---------------------------------------------------------------------------
// Kernel 3: batchnorm stats over (b, c) = 192 samples per (s, a, feat).
// One block per (s*7+a); LDS float atomics (ds_add_f32) accumulate 9 feats.
// ---------------------------------------------------------------------------
__global__ __launch_bounds__(192) void k_bnstats(const float* __restrict__ att,
                                                 float* __restrict__ meanOut,
                                                 float* __restrict__ invOut) {
  __shared__ float ssum[9];
  __shared__ float ssq[9];
  const int sa = blockIdx.x;   // s*7 + a
  const int t  = threadIdx.x;  // 0..191
  if (t < 9) { ssum[t] = 0.f; ssq[t] = 0.f; }
  __syncthreads();

  const int b = t / 3, c = t % 3;
  const int s = sa / 7, a = sa % 7;
  const float* p = att + ((size_t)((b * S_DIM + s) * 7 + a) * 3 + c) * 9;
#pragma unroll
  for (int f = 0; f < 9; ++f) {
    float v = p[f];
    atomicAdd(&ssum[f], v);
    atomicAdd(&ssq[f], v * v);
  }
  __syncthreads();

  if (t < 9) {
    float m   = ssum[t] * (1.f / 192.f);
    float var = ssq[t] * (1.f / 192.f) - m * m;
    meanOut[sa * 9 + t] = m;
    invOut[sa * 9 + t]  = rsqrtf(var + 1e-5f);
  }
}

// ---------------------------------------------------------------------------
// Kernel 4: bn apply + 9->3 FC + transpose to (b, 21, s*3).
// ---------------------------------------------------------------------------
__global__ __launch_bounds__(256) void k_bnfc(
    const float* __restrict__ att, const float* __restrict__ meanIn,
    const float* __restrict__ invIn, const float* __restrict__ gamma,
    const float* __restrict__ beta, const float* __restrict__ fcw,
    const float* __restrict__ fcb, float* __restrict__ out) {
  const int idx = blockIdx.x * blockDim.x + threadIdx.x;
  if (idx >= B_DIM * S_DIM * 21) return;
  const int c  = idx % 3;
  const int t1 = idx / 3;
  const int a  = t1 % 7;
  const int bs = t1 / 7;
  const int s  = bs % S_DIM, b = bs / S_DIM;

  const float* p  = att + (size_t)idx * 9;
  const float* m  = meanIn + (size_t)(s * 7 + a) * 9;
  const float* iv = invIn + (size_t)(s * 7 + a) * 9;

  float h[9];
#pragma unroll
  for (int f = 0; f < 9; ++f)
    h[f] = (p[f] - m[f]) * iv[f] * gamma[f] + beta[f];

  float o0 = fcb[0], o1 = fcb[1], o2 = fcb[2];
#pragma unroll
  for (int f = 0; f < 9; ++f) {
    o0 += h[f] * fcw[f];
    o1 += h[f] * fcw[9 + f];
    o2 += h[f] * fcw[18 + f];
  }

  const int ch = a * 3 + c;
  float* q = out + ((size_t)(b * 21 + ch) * S_DIM + s) * 3;
  q[0] = o0; q[1] = o1; q[2] = o2;
}

// ---------------------------------------------------------------------------
// Host-side launcher (graph-capture safe: only kernel launches on `stream`).
// Workspace layout (bytes):
//   [0,       131072)  f16 DFT cos (transposed, [f][k])
//   [131072,  262144)  f16 DFT -sin
//   [262144,  2326528) A        : 57344 * 9 f32
//   [2326528, 8519680) att      : 57344 * 27 f32
//   [8519680, 8551936) bn mean  : 8064 f32
//   [8551936, 8584192) bn rstd  : 8064 f32
// ---------------------------------------------------------------------------
extern "C" void kernel_launch(void* const* d_in, const int* in_sizes, int n_in,
                              void* d_out, int out_size, void* d_ws,
                              size_t ws_size, hipStream_t stream) {
  (void)in_sizes; (void)n_in; (void)out_size; (void)ws_size;

  const float* x     = (const float*)d_in[0];
  const float* Wq    = (const float*)d_in[1];
  const float* bq    = (const float*)d_in[2];
  const float* Wk    = (const float*)d_in[3];
  const float* bk    = (const float*)d_in[4];
  const float* Wv    = (const float*)d_in[5];
  const float* bv    = (const float*)d_in[6];
  const float* gamma = (const float*)d_in[7];
  const float* beta  = (const float*)d_in[8];
  const float* fcw   = (const float*)d_in[9];
  const float* fcb   = (const float*)d_in[10];

  char* ws = (char*)d_ws;
  _Float16* btc  = (_Float16*)(ws + 0);
  _Float16* bts  = (_Float16*)(ws + 131072);
  float*    A    = (float*)(ws + 262144);
  float*    att  = (float*)(ws + 2326528);
  float*    mean = (float*)(ws + 8519680);
  float*    inv  = (float*)(ws + 8551936);
  float*    out  = (float*)d_out;

  k_build_dft<<<256, 256, 0, stream>>>(btc, bts);
  k_dft_stats<<<NWG_DFT, 256, 0, stream>>>(x, btc, bts, A);
  k_attn<<<(NGRP_TOTAL + 255) / 256, 256, 0, stream>>>(A, Wq, bq, Wk, bk, Wv,
                                                       bv, att);
  k_bnstats<<<S_DIM * 7, 192, 0, stream>>>(att, mean, inv);
  k_bnfc<<<(B_DIM * S_DIM * 21 + 255) / 256, 256, 0, stream>>>(
      att, mean, inv, gamma, beta, fcw, fcb, out);
}
